// DopamineArea_86079734546869
// MI455X (gfx1250) — compile-verified, matrix operand order checked by reading, not run on hardware
//
#include <hip/hip_runtime.h>

// ---------------------------------------------------------------------------
// MI455X (gfx1250) DopamineArea forward:
//   out_spikes = (spikes >= 0.5) ? 1 : 0
//   S          = mean(out_spikes)
//   delta      = S - P
//
// Pure streaming problem: 512 MB minimum traffic @ 23.3 TB/s ~= 22 us.
// Kernel 1 reads spikes once, writes out_spikes, and produces the exact
// integer spike count (WMMA-based intra-wave reduction + u32 atomics).
// Kernel 2 reads P once and writes delta = S - P.
// Non-temporal hints everywhere: nothing is re-read, don't pollute L2.
// ---------------------------------------------------------------------------

typedef __attribute__((ext_vector_type(4)))  float    f4;
typedef __attribute__((ext_vector_type(16))) _Float16 v16h;
typedef __attribute__((ext_vector_type(8)))  float    v8f;

#define THRESH   0.5f
#define BLOCK    256
#define GRID     4096   // 1,048,576 lanes -> 8 f4/lane for n=2^25 (f16-exact counts)

// ds_swizzle group-of-32 SWAPX16: xor_mask=0x10, or_mask=0, and_mask=0x1f
#define SWIZZLE_SWAPX16 0x401F

__device__ __forceinline__ float swap16_add(float v) {
    int other = __builtin_amdgcn_ds_swizzle(__float_as_int(v), SWIZZLE_SWAPX16);
    return v + __int_as_float(other);
}

// ---------------------------------------------------------------------------
// Kernel 1: spikes -> out_spikes bits + exact global spike count.
// Wave reduction trick: each lane puts its count into one f16 slot of the
// 16x32 A matrix (every lane owns a distinct (row,K) slot per the CDNA5 A
// layout), B = all-ones 32x16.  D[m][n] = count_m + count_{m+16}; each lane
// sums its 8 accumulator VGPRs (rows 0-7 or 8-15 of one column), and a single
// SWAPX16 swizzle-add with the partner half gives the full 32-lane total.
// All quantities are small exact integers, so f16/f32 arithmetic is exact.
// ---------------------------------------------------------------------------
__global__ __launch_bounds__(BLOCK) void spike_count_kernel(
    const float* __restrict__ spikes,
    float*       __restrict__ out_spikes,
    unsigned int* __restrict__ count,
    long long n)
{
    const long long n4     = n >> 2;
    const long long tid    = (long long)blockIdx.x * blockDim.x + threadIdx.x;
    const long long stride = (long long)gridDim.x * blockDim.x;

    const f4* __restrict__ s4 = (const f4*)spikes;
    f4*       __restrict__ o4 = (f4*)out_spikes;

    unsigned int cnt = 0;

    // Main vectorized stream: b128 NT loads, b128 NT stores.
    // Unroll 4 -> 4 outstanding b128 loads per wave before the stores.
#pragma unroll 4
    for (long long i = tid; i < n4; i += stride) {
        f4 s = __builtin_nontemporal_load(&s4[i]);
        unsigned c0 = (s.x >= THRESH);
        unsigned c1 = (s.y >= THRESH);
        unsigned c2 = (s.z >= THRESH);
        unsigned c3 = (s.w >= THRESH);
        f4 b;
        b.x = (float)c0; b.y = (float)c1; b.z = (float)c2; b.w = (float)c3;
        __builtin_nontemporal_store(b, &o4[i]);
        cnt += c0 + c1 + c2 + c3;
    }
    // Scalar tail (n = 2^25 here so this never runs, but stay generic).
    for (long long i = (n4 << 2) + tid; i < n; i += stride) {
        float s = spikes[i];
        unsigned c = (s >= THRESH);
        out_spikes[i] = (float)c;
        cnt += c;
    }

    // --- WMMA intra-wave reduction of 32 lane counts -----------------------
    v16h a = {};                       // one slot per lane, rest zero
    a[0] = (_Float16)(float)cnt;       // exact: cnt <= 2048 by grid sizing
    v16h ones;
#pragma unroll
    for (int k = 0; k < 16; ++k) ones[k] = (_Float16)1.0f;
    v8f c = {};
    // D = A x ones + 0 : row m of D replicates count_m + count_{m+16}
    c = __builtin_amdgcn_wmma_f32_16x16x32_f16(
            /*neg_a=*/false, a, /*neg_b=*/false, ones,
            /*c_mod=*/(short)0, c, /*reuse_a=*/false, /*reuse_b=*/false);

    float local = 0.0f;
#pragma unroll
    for (int k = 0; k < 8; ++k) local += c[k];     // rows 0-7 (lanes<16) or 8-15
    float tot = swap16_add(local);                 // + complementary half

    if ((threadIdx.x & 31u) == 0u)
        atomicAdd(count, (unsigned int)(tot + 0.5f));   // exact integer total
}

// ---------------------------------------------------------------------------
// Kernel 2: delta = S - P.  Counter address is uniform -> scalar load.
// ---------------------------------------------------------------------------
__global__ __launch_bounds__(BLOCK) void delta_kernel(
    const float* __restrict__ P,
    float*       __restrict__ delta,
    const unsigned int* __restrict__ count,
    long long n)
{
    const float S = (float)(*count) / (float)n;   // single rounding, like jnp.mean

    const long long n4     = n >> 2;
    const long long tid    = (long long)blockIdx.x * blockDim.x + threadIdx.x;
    const long long stride = (long long)gridDim.x * blockDim.x;

    const f4* __restrict__ p4 = (const f4*)P;
    f4*       __restrict__ d4 = (f4*)delta;

#pragma unroll 4
    for (long long i = tid; i < n4; i += stride) {
        f4 p = __builtin_nontemporal_load(&p4[i]);
        f4 d;
        d.x = S - p.x; d.y = S - p.y; d.z = S - p.z; d.w = S - p.w;
        __builtin_nontemporal_store(d, &d4[i]);
    }
    for (long long i = (n4 << 2) + tid; i < n; i += stride)
        delta[i] = S - P[i];
}

// ---------------------------------------------------------------------------
// kernel_launch: d_out = [delta (n floats) | out_spikes (n floats)]
// ---------------------------------------------------------------------------
extern "C" void kernel_launch(void* const* d_in, const int* in_sizes, int n_in,
                              void* d_out, int out_size, void* d_ws, size_t ws_size,
                              hipStream_t stream) {
    const float* spikes = (const float*)d_in[0];
    const float* P      = (const float*)d_in[1];
    float* out          = (float*)d_out;

    const long long n = (long long)in_sizes[0];
    float* delta      = out;          // first output in return order
    float* out_spikes = out + n;      // second output

    unsigned int* count = (unsigned int*)d_ws;

    // Zero the exact-count accumulator (graph-capturable async memset).
    hipMemsetAsync(count, 0, sizeof(unsigned int), stream);

    spike_count_kernel<<<GRID, BLOCK, 0, stream>>>(spikes, out_spikes, count, n);
    delta_kernel<<<GRID, BLOCK, 0, stream>>>(P, delta, count, n);
}
